// GraphConvNet_actor_4415226380547
// MI455X (gfx1250) — matrix operand order repaired and checked by reading, not compiled, not run on hardware
//
#include <hip/hip_runtime.h>
#include <hip/hip_bf16.h>

typedef __attribute__((ext_vector_type(16))) __bf16 v16bf;
typedef __attribute__((ext_vector_type(8)))  float  v8f;

#define ZDIM   154
#define ZPAD   160
#define KSTEPS (ZPAD / 32)     // 5
#define LDIM   64
#define TILE_M 16
#define CWAVES 4
#define EDGES_PER_BLOCK (CWAVES * TILE_M)

__device__ __forceinline__ float leaky_f(float x)   { return x > 0.f ? x : 0.01f * x; }
__device__ __forceinline__ float sigmoid_f(float x) { return 1.f / (1.f + __expf(-x)); }
// branchless softplus: max(x,0) + log1p(exp(-|x|)); exp underflow handles tails
__device__ __forceinline__ float softplus_f(float x)
{
    return fmaxf(x, 0.f) + __logf(1.f + __expf(-fabsf(x)));
}

// Deterministic single-instruction f32 atomic scatter (device scope: edges from
// different WGPs accumulate into the same node row).
__device__ __forceinline__ void atomic_add_f32(float* p, float v)
{
    asm volatile("global_atomic_add_f32 %0, %1, off scope:SCOPE_DEV"
                 :: "v"(p), "v"(v) : "memory");
}

// Inverse fragment-layout swizzles (ISA 7.12.2, 16-bit operands, wave32):
//   A (16xK per-lane layout): k_in_step = i + (i&8) + 8*hi
//   B (Kx16 per-lane layout): k_in_step = i + (i&8) + 8*hi
// We store tiles in LDS already permuted so each lane reads its 16 halfs
// (32 bytes) contiguously -> 2x ds_load_b128 per fragment.
__device__ __forceinline__ void a_swz(int kk, int& hi, int& i)
{
    hi = (kk >> 3) & 1;
    i  = (kk & 7) + ((kk & 16) >> 1);
}
__device__ __forceinline__ void b_swz(int kk, int& hi, int& i)
{
    hi = (kk >> 3) & 1;
    i  = (kk & 7) + ((kk & 16) >> 1);
}

// ---------------------------------------------------------------------------
// CGConv edge kernel: per wave, one 16-edge x 64-channel tile via bf16 WMMA.
//   z_e = [h[dst] | h[src] | gauss(len)]  (154 -> padded 160)
//   msg = sigmoid(z@Wf + bf) * softplus(z@Ws + bs)   -> atomic scatter to dst
// ---------------------------------------------------------------------------
__global__ __launch_bounds__(CWAVES * 32)
void cgconv_edge_kernel(const float* __restrict__ h,
                        const int*   __restrict__ src,
                        const int*   __restrict__ dst,
                        const float* __restrict__ elen,
                        const float* __restrict__ wf,
                        const float* __restrict__ bfv,
                        const float* __restrict__ wsw,
                        const float* __restrict__ bsv,
                        float* __restrict__ out,
                        int E)
{
    // Swizzled fragment storage.
    //  Weights: [ks][hi][n][i] -> ((ks*2+hi)*64 + n)*16 + i      (20 KB each)
    //  Z tile : [ks][hi][m][i] -> ((ks*2+hi)*16 + m)*16 + i      (5 KB/wave)
    __shared__ __align__(32) __bf16 sWf[KSTEPS * 2 * LDIM * 16];
    __shared__ __align__(32) __bf16 sWs[KSTEPS * 2 * LDIM * 16];
    __shared__ __align__(32) __bf16 sZ[CWAVES][KSTEPS * 2 * TILE_M * 16];
    __shared__ float sBf[LDIM], sBs[LDIM];
    __shared__ int   sDst[CWAVES][TILE_M];

    const int tid = threadIdx.x;

    // Stage both weight matrices into LDS (bf16, fragment-swizzled, K padded).
    for (int idx = tid; idx < ZPAD * LDIM; idx += blockDim.x) {
        int k = idx >> 6, j = idx & 63;            // k in [0,160), j in [0,64)
        float vf = (k < ZDIM) ? wf[k * LDIM + j]  : 0.f;
        float vs = (k < ZDIM) ? wsw[k * LDIM + j] : 0.f;
        int ks = k >> 5, kk = k & 31, hi, i;
        b_swz(kk, hi, i);
        int a = ((ks * 2 + hi) * LDIM + j) * 16 + i;
        sWf[a] = (__bf16)vf;
        sWs[a] = (__bf16)vs;
    }
    for (int i = tid; i < LDIM; i += blockDim.x) { sBf[i] = bfv[i]; sBs[i] = bsv[i]; }
    __syncthreads();

    const int wave  = tid >> 5;
    const int lane  = tid & 31;
    const int row   = lane & 15;
    const int hi    = lane >> 4;
    const int ebase = blockIdx.x * EDGES_PER_BLOCK + wave * TILE_M;

    if (lane < TILE_M) {
        int e = ebase + lane;
        sDst[wave][lane] = (e < E) ? dst[e] : -1;
    }

    // Build the z tile (16 edges x 160, bf16, A-fragment-swizzled).
    for (int m = 0; m < TILE_M; ++m) {
        int e = ebase + m;
        bool valid = (e < E);
        int sI = 0, dI = 0; float len = 0.f;
        if (valid) { sI = src[e]; dI = dst[e]; len = elen[e]; }
        for (int k = lane; k < ZPAD; k += 32) {
            float v = 0.f;
            if (valid) {
                if (k < LDIM)            v = h[(size_t)dI * LDIM + k];
                else if (k < 2 * LDIM)   v = h[(size_t)sI * LDIM + (k - LDIM)];
                else if (k < ZDIM) {
                    float mu = 0.2f * (float)(k - 2 * LDIM);   // linspace(0,5,26)
                    float t  = len - mu;
                    v = __expf(-25.f * t * t);                 // exp(-(d-mu)^2/0.04)
                }
            }
            int ks = k >> 5, kk = k & 31, zh, zi;
            a_swz(kk, zh, zi);
            sZ[wave][((ks * 2 + zh) * TILE_M + m) * 16 + zi] = (__bf16)v;
        }
    }

    v8f zero = {0.f, 0.f, 0.f, 0.f, 0.f, 0.f, 0.f, 0.f};
    v8f accF[4] = {zero, zero, zero, zero};
    v8f accS[4] = {zero, zero, zero, zero};

    const v16bf* zfrag  = (const v16bf*)&sZ[wave][0];
    const v16bf* wffrag = (const v16bf*)sWf;
    const v16bf* wsfrag = (const v16bf*)sWs;

    #pragma unroll
    for (int ks = 0; ks < KSTEPS; ++ks) {
        v16bf a = zfrag[(ks * 2 + hi) * TILE_M + row];          // 2x ds_load_b128
        #pragma unroll
        for (int nt = 0; nt < 4; ++nt) {
            int fb = (ks * 2 + hi) * LDIM + nt * 16 + row;
            v16bf bF = wffrag[fb];                              // 2x ds_load_b128
            v16bf bS = wsfrag[fb];                              // 2x ds_load_b128
            accF[nt] = __builtin_amdgcn_wmma_f32_16x16x32_bf16(
                false, a, false, bF, (short)0, accF[nt], false, false);
            accS[nt] = __builtin_amdgcn_wmma_f32_16x16x32_bf16(
                false, a, false, bS, (short)0, accS[nt], false, false);
        }
    }

    // Epilogue: f32 gating + atomic scatter. D layout: VGPR r -> M=r+8*hi, N=lane%16.
    #pragma unroll
    for (int nt = 0; nt < 4; ++nt) {
        int n = nt * 16 + row;
        float bF = sBf[n], bS = sBs[n];
        #pragma unroll
        for (int r = 0; r < 8; ++r) {
            int m = r + hi * 8;
            int dnode = sDst[wave][m];
            if (dnode >= 0) {
                float g = sigmoid_f(accF[nt][r] + bF) * softplus_f(accS[nt][r] + bS);
                atomic_add_f32(&out[(size_t)dnode * LDIM + n], g);
            }
        }
    }
}

// ---------------------------------------------------------------------------
// Node-side kernels (small FLOP share; exact f32)
// ---------------------------------------------------------------------------
__global__ void embed1_kernel(const float* __restrict__ x, const float* __restrict__ fs,
                              const float* __restrict__ fn, const float* __restrict__ w,
                              const float* __restrict__ b, float* __restrict__ out, int N)
{
    int n = blockIdx.x * blockDim.x + threadIdx.x;
    if (n >= N) return;
    float in[8];
    in[0] = x[n * 4 + 0]; in[1] = x[n * 4 + 1]; in[2] = x[n * 4 + 2]; in[3] = x[n * 4 + 3];
    in[4] = fs[n * 3 + 0]; in[5] = fs[n * 3 + 1]; in[6] = fs[n * 3 + 2];
    in[7] = fn[n];
    for (int j = 0; j < 128; ++j) {
        float acc = b[j];
        #pragma unroll
        for (int k = 0; k < 8; ++k) acc = fmaf(in[k], w[k * 128 + j], acc);
        out[(size_t)n * 128 + j] = leaky_f(acc);
    }
}

template <int IN, int OUT>
__global__ void dense_leaky_kernel(const float* __restrict__ in, const float* __restrict__ w,
                                   const float* __restrict__ b, float* __restrict__ out, int N)
{
    int n = blockIdx.x * blockDim.x + threadIdx.x;
    if (n >= N) return;
    float r[IN];
    #pragma unroll
    for (int k = 0; k < IN; ++k) r[k] = in[(size_t)n * IN + k];
    for (int j = 0; j < OUT; ++j) {
        float acc = b[j];
        #pragma unroll
        for (int k = 0; k < IN; ++k) acc = fmaf(r[k], w[k * OUT + j], acc);
        out[(size_t)n * OUT + j] = leaky_f(acc);
    }
}

__global__ void desc_kernel(const float* __restrict__ in,
                            const float* __restrict__ w1, const float* __restrict__ b1,
                            const float* __restrict__ w2, const float* __restrict__ b2,
                            float* __restrict__ out, int N)
{
    int n = blockIdx.x * blockDim.x + threadIdx.x;
    if (n >= N) return;
    float r[64];
    #pragma unroll
    for (int k = 0; k < 64; ++k) r[k] = in[(size_t)n * 64 + k];
    float hbuf[64];
    for (int j = 0; j < 64; ++j) {
        float acc = b1[j];
        #pragma unroll
        for (int k = 0; k < 64; ++k) acc = fmaf(r[k], w1[k * 64 + j], acc);
        hbuf[j] = fmaxf(acc, 0.f);
    }
    #pragma unroll
    for (int j = 0; j < 3; ++j) {
        float acc = b2[j];
        #pragma unroll
        for (int k = 0; k < 64; ++k) acc = fmaf(hbuf[k], w2[k * 3 + j], acc);
        out[(size_t)n * 3 + j] = acc;
    }
}

__global__ void leaky_kernel(float* __restrict__ p, int n)
{
    int i = blockIdx.x * blockDim.x + threadIdx.x;
    if (i < n) p[i] = leaky_f(p[i]);
}

// ---------------------------------------------------------------------------
extern "C" void kernel_launch(void* const* d_in, const int* in_sizes, int n_in,
                              void* d_out, int out_size, void* d_ws, size_t ws_size,
                              hipStream_t stream)
{
    const float* x   = (const float*)d_in[0];
    const float* fs  = (const float*)d_in[1];
    const float* fn  = (const float*)d_in[2];
    const int*   ei  = (const int*)  d_in[3];
    const float* el  = (const float*)d_in[4];
    const float* ew1 = (const float*)d_in[5];  const float* eb1 = (const float*)d_in[6];
    const float* ew2 = (const float*)d_in[7];  const float* eb2 = (const float*)d_in[8];
    const float* ew3 = (const float*)d_in[9];  const float* eb3 = (const float*)d_in[10];
    const float* cwf[3]; const float* cbf[3]; const float* cws[3]; const float* cbs[3];
    for (int c = 0; c < 3; ++c) {
        cwf[c] = (const float*)d_in[11 + 4 * c + 0];
        cbf[c] = (const float*)d_in[11 + 4 * c + 1];
        cws[c] = (const float*)d_in[11 + 4 * c + 2];
        cbs[c] = (const float*)d_in[11 + 4 * c + 3];
    }
    const float* dw1 = (const float*)d_in[23]; const float* db1 = (const float*)d_in[24];
    const float* dw2 = (const float*)d_in[25]; const float* db2 = (const float*)d_in[26];
    float* outp = (float*)d_out;

    const int N = in_sizes[0] / 4;
    const int E = in_sizes[4];
    const int* srcI = ei;        // edge_index[0]
    const int* dstI = ei + E;    // edge_index[1]

    float* h1 = (float*)d_ws;                 // N*128
    float* h2 = h1 + (size_t)N * 128;         // N*128
    float* hA = h1;                           // reused as N*64 after embed3
    float* hB = h2;

    const int nb  = (N + 255) / 256;
    const int ebk = (E + EDGES_PER_BLOCK - 1) / EDGES_PER_BLOCK;
    const int nl  = (N * 64 + 255) / 256;
    const size_t hbytes = (size_t)N * 64 * sizeof(float);

    // Embedding MLP: 8 -> 128 -> 128 -> 64, leaky
    embed1_kernel<<<nb, 256, 0, stream>>>(x, fs, fn, ew1, eb1, h1, N);
    dense_leaky_kernel<128, 128><<<nb, 256, 0, stream>>>(h1, ew2, eb2, h2, N);
    dense_leaky_kernel<128, 64><<<nb, 256, 0, stream>>>(h2, ew3, eb3, hA, N);

    // conv1: out = leaky(h + scatter(msg))
    hipMemcpyAsync(hB, hA, hbytes, hipMemcpyDeviceToDevice, stream);
    cgconv_edge_kernel<<<ebk, CWAVES * 32, 0, stream>>>(hA, srcI, dstI, el,
                                                        cwf[0], cbf[0], cws[0], cbs[0], hB, E);
    leaky_kernel<<<nl, 256, 0, stream>>>(hB, N * 64);

    // conv2
    hipMemcpyAsync(hA, hB, hbytes, hipMemcpyDeviceToDevice, stream);
    cgconv_edge_kernel<<<ebk, CWAVES * 32, 0, stream>>>(hB, srcI, dstI, el,
                                                        cwf[1], cbf[1], cws[1], cbs[1], hA, E);
    leaky_kernel<<<nl, 256, 0, stream>>>(hA, N * 64);

    // conv3 (no leaky)
    hipMemcpyAsync(hB, hA, hbytes, hipMemcpyDeviceToDevice, stream);
    cgconv_edge_kernel<<<ebk, CWAVES * 32, 0, stream>>>(hA, srcI, dstI, el,
                                                        cwf[2], cbf[2], cws[2], cbs[2], hB, E);

    // Descriptor head: 64 -> relu 64 -> 3
    desc_kernel<<<nb, 256, 0, stream>>>(hB, dw1, db1, dw2, db2, outp, N);
}
